// NLITBlock_74560632259058
// MI455X (gfx1250) — matrix operand order repaired and checked by reading, not compiled, not run on hardware
//
#include <hip/hip_runtime.h>
#include <math.h>

typedef _Float16 f16;
typedef __attribute__((ext_vector_type(16))) _Float16 v16h;
typedef __attribute__((ext_vector_type(8)))  _Float16 v8h;
typedef __attribute__((ext_vector_type(2)))  _Float16 h2v;
typedef __attribute__((ext_vector_type(8)))  float    v8f;

constexpr int NB = 8, NC = 64, NO = 64, NH = 192, NW = 192, NM1 = 48, NM2 = 48;
constexpr float EPSV = 1e-5f;

__device__ __forceinline__ float gelu_f(float x) {
  return 0.5f * x * (1.0f + erff(x * 0.7071067811865476f));
}

// ---------------------------------------------------------------------------
// Batched WMMA GEMM:  D[m,n] = alpha * sum_k A[m,k]*Bt[n,k] + beta*D
// A,Bt: f16, K contiguous.  D: f32, N contiguous.  Two batch dims w/ strides.
// K multiple of 32 (operands zero-padded).  Three variants:
//   wmma_gemm16  : 1 wave,  16x16 tile   (fallback, any M/16, N/16)
//   wmma_gemm32  : 1 wave,  32x32 tile   (M%32==0, N%32==0) -> 2x fragment reuse
//   wmma_gemm_mlp: 4 waves, 128x32 tile, B staged in LDS (M%128==0, N%32==0)
// ---------------------------------------------------------------------------
struct GemmP {
  const f16* A; const f16* Bt; float* D;
  int M, N, K;
  int lda, ldb, ldc;
  long long sA1, sA2, sB1, sB2, sD1, sD2;
  int B2;
  float alpha, beta;
};

// A fragment 16x32 f16: lane row = lo, half hl; VGPR j holds K pair
// k = 16*hl + 2*j   (contiguous, matches B fragment)
__device__ __forceinline__ v16h load_afrag(const f16* Arow, int k, int hl) {
  v16h a;
#pragma unroll
  for (int j = 0; j < 8; ++j) {
    const int kk = k + (hl << 4) + (j << 1);
    h2v pr = *(const h2v*)(Arow + kk);
    a[2 * j]     = pr.x;
    a[2 * j + 1] = pr.y;
  }
  return a;
}

// B fragment 32x16 f16: lane = column; lanes 0-15 K=0..15, 16-31 K=16..31.
// ptr must point at the 16 contiguous halves for this lane.
__device__ __forceinline__ v16h load_bfrag(const f16* ptr) {
  v8h b0v = *(const v8h*)(ptr);
  v8h b1v = *(const v8h*)(ptr + 8);
  v16h b;
#pragma unroll
  for (int i = 0; i < 8; ++i) { b[i] = b0v[i]; b[8 + i] = b1v[i]; }
  return b;
}

__device__ __forceinline__ void store_tile(float* Dbase, const v8f& acc,
                                           int m0, int n0, int ldc,
                                           float alpha, float beta,
                                           int lo, int hl) {
  float* q = Dbase + (n0 + lo);
  const int mr = m0 + (hl << 3);
#pragma unroll
  for (int r = 0; r < 8; ++r) {
    const size_t idx = (size_t)(mr + r) * ldc;
    float v = alpha * acc[r];
    if (beta != 0.0f) v += beta * q[idx];
    q[idx] = v;
  }
}

__global__ __launch_bounds__(32) void wmma_gemm16(GemmP p) {
  const int lane = threadIdx.x;
  const int lo = lane & 15, hl = lane >> 4;
  const int n0 = blockIdx.x << 4;
  const int m0 = blockIdx.y << 4;
  const int bb = blockIdx.z;
  const int b1 = bb / p.B2, b2 = bb - b1 * p.B2;

  const f16* Arow = p.A + (size_t)b1 * p.sA1 + (size_t)b2 * p.sA2
                        + (size_t)(m0 + lo) * p.lda;
  const f16* Brow = p.Bt + (size_t)b1 * p.sB1 + (size_t)b2 * p.sB2
                         + (size_t)(n0 + lo) * p.ldb + (hl << 4);
  v8f acc = {};
  for (int k = 0; k < p.K; k += 32) {
    __builtin_prefetch(Arow + k + 32, 0, 0);
    __builtin_prefetch(Brow + k + 32, 0, 0);
    v16h a = load_afrag(Arow, k, hl);
    v16h b = load_bfrag(Brow + k);
    acc = __builtin_amdgcn_wmma_f32_16x16x32_f16(false, a, false, b,
                                                 (short)0, acc, false, false);
  }
  float* Dp = p.D + (size_t)b1 * p.sD1 + (size_t)b2 * p.sD2;
  store_tile(Dp, acc, m0, n0, p.ldc, p.alpha, p.beta, lo, hl);
}

__global__ __launch_bounds__(32) void wmma_gemm32(GemmP p) {
  const int lane = threadIdx.x;
  const int lo = lane & 15, hl = lane >> 4;
  const int n0 = blockIdx.x << 5;
  const int m0 = blockIdx.y << 5;
  const int bb = blockIdx.z;
  const int b1 = bb / p.B2, b2 = bb - b1 * p.B2;

  const f16* A0 = p.A + (size_t)b1 * p.sA1 + (size_t)b2 * p.sA2
                      + (size_t)(m0 + lo) * p.lda;
  const f16* A1 = A0 + (size_t)16 * p.lda;
  const f16* B0 = p.Bt + (size_t)b1 * p.sB1 + (size_t)b2 * p.sB2
                       + (size_t)(n0 + lo) * p.ldb + (hl << 4);
  const f16* B1 = B0 + (size_t)16 * p.ldb;

  v8f a00 = {}, a01 = {}, a10 = {}, a11 = {};
  for (int k = 0; k < p.K; k += 32) {
    __builtin_prefetch(A0 + k + 32, 0, 0);
    __builtin_prefetch(B0 + k + 32, 0, 0);
    v16h fa0 = load_afrag(A0, k, hl);
    v16h fa1 = load_afrag(A1, k, hl);
    v16h fb0 = load_bfrag(B0 + k);
    v16h fb1 = load_bfrag(B1 + k);
    a00 = __builtin_amdgcn_wmma_f32_16x16x32_f16(false, fa0, false, fb0, (short)0, a00, false, false);
    a01 = __builtin_amdgcn_wmma_f32_16x16x32_f16(false, fa0, false, fb1, (short)0, a01, false, false);
    a10 = __builtin_amdgcn_wmma_f32_16x16x32_f16(false, fa1, false, fb0, (short)0, a10, false, false);
    a11 = __builtin_amdgcn_wmma_f32_16x16x32_f16(false, fa1, false, fb1, (short)0, a11, false, false);
  }
  float* Dp = p.D + (size_t)b1 * p.sD1 + (size_t)b2 * p.sD2;
  store_tile(Dp, a00, m0,      n0,      p.ldc, p.alpha, p.beta, lo, hl);
  store_tile(Dp, a01, m0,      n0 + 16, p.ldc, p.alpha, p.beta, lo, hl);
  store_tile(Dp, a10, m0 + 16, n0,      p.ldc, p.alpha, p.beta, lo, hl);
  store_tile(Dp, a11, m0 + 16, n0 + 16, p.ldc, p.alpha, p.beta, lo, hl);
}

// 4 waves stacked in M (128x32 block tile); shared 32x32 B tile staged in LDS.
__global__ __launch_bounds__(128) void wmma_gemm_mlp(GemmP p) {
  __shared__ __align__(16) f16 Bs[32 * 32];   // [n 0..31][k 0..31]
  const int t = threadIdx.x;
  const int lane = t & 31, wv = t >> 5;
  const int lo = lane & 15, hl = lane >> 4;
  const int n0 = blockIdx.x << 5;
  const int m0 = (blockIdx.y << 7) + (wv << 5);
  const int bb = blockIdx.z;
  const int b1 = bb / p.B2, b2 = bb - b1 * p.B2;

  const f16* A0 = p.A + (size_t)b1 * p.sA1 + (size_t)b2 * p.sA2
                      + (size_t)(m0 + lo) * p.lda;
  const f16* A1 = A0 + (size_t)16 * p.lda;
  // cooperative B tile fetch: thread t -> row n0 + (t>>2), halves [(t&3)*8, +8)
  const f16* Bg = p.Bt + (size_t)b1 * p.sB1 + (size_t)b2 * p.sB2
                       + (size_t)(n0 + (t >> 2)) * p.ldb + ((t & 3) << 3);
  f16* Bsw = &Bs[(t >> 2) * 32 + ((t & 3) << 3)];
  const f16* Bl0 = &Bs[lo * 32 + (hl << 4)];
  const f16* Bl1 = &Bs[(16 + lo) * 32 + (hl << 4)];

  v8f a00 = {}, a01 = {}, a10 = {}, a11 = {};
  for (int k = 0; k < p.K; k += 32) {
    __syncthreads();                     // all waves done reading previous tile
    *(v8h*)Bsw = *(const v8h*)(Bg + k);  // stage 32x32 B tile (2 KB)
    __syncthreads();
    __builtin_prefetch(A0 + k + 32, 0, 0);
    __builtin_prefetch(A1 + k + 32, 0, 0);
    v16h fa0 = load_afrag(A0, k, hl);
    v16h fa1 = load_afrag(A1, k, hl);
    v16h fb0 = load_bfrag(Bl0);          // ds_load_b128 x2
    v16h fb1 = load_bfrag(Bl1);
    a00 = __builtin_amdgcn_wmma_f32_16x16x32_f16(false, fa0, false, fb0, (short)0, a00, false, false);
    a01 = __builtin_amdgcn_wmma_f32_16x16x32_f16(false, fa0, false, fb1, (short)0, a01, false, false);
    a10 = __builtin_amdgcn_wmma_f32_16x16x32_f16(false, fa1, false, fb0, (short)0, a10, false, false);
    a11 = __builtin_amdgcn_wmma_f32_16x16x32_f16(false, fa1, false, fb1, (short)0, a11, false, false);
  }
  float* Dp = p.D + (size_t)b1 * p.sD1 + (size_t)b2 * p.sD2;
  store_tile(Dp, a00, m0,      n0,      p.ldc, p.alpha, p.beta, lo, hl);
  store_tile(Dp, a01, m0,      n0 + 16, p.ldc, p.alpha, p.beta, lo, hl);
  store_tile(Dp, a10, m0 + 16, n0,      p.ldc, p.alpha, p.beta, lo, hl);
  store_tile(Dp, a11, m0 + 16, n0 + 16, p.ldc, p.alpha, p.beta, lo, hl);
}

// ---------------------------------------------------------------------------
// Elementwise / staging kernels
// ---------------------------------------------------------------------------

// dst[r*Kp + k] = (k < Ks) ? src[k*sld + r] : 0   (transpose-cast-pad)
__global__ void k_tcast(const float* src, f16* dst, int R, int Kp, int Ks, int sld) {
  int i = blockIdx.x * blockDim.x + threadIdx.x;
  if (i >= R * Kp) return;
  int k = i % Kp, r = i / Kp;
  dst[i] = (k < Ks) ? (f16)src[(size_t)k * sld + r] : (f16)0.0f;
}

// dst[r*Kp + k] = (k < Ks) ? src[r*sld + k] : 0   (cast-pad, no transpose)
__global__ void k_ccast(const float* src, f16* dst, int R, int Kp, int Ks, int sld) {
  int i = blockIdx.x * blockDim.x + threadIdx.x;
  if (i >= R * Kp) return;
  int k = i % Kp, r = i / Kp;
  dst[i] = (k < Ks) ? (f16)src[(size_t)r * sld + k] : (f16)0.0f;
}

__global__ void k_cast(const float* src, f16* dst, int n) {
  int i = blockIdx.x * blockDim.x + threadIdx.x;
  if (i < n) dst[i] = (f16)src[i];
}

// Xw[(b,h,w), 0:96] = {x[b,:,h,w], wc[w], zeros}
__global__ void k_build_xw(const float* x, f16* Xw) {
  int p = blockIdx.x * blockDim.x + threadIdx.x;
  if (p >= NB * NH * NW) return;
  int w = p % NW, h = (p / NW) % NH, b = p / (NH * NW);
  f16* row = Xw + (size_t)p * 96;
#pragma unroll 4
  for (int c = 0; c < 64; ++c)
    row[c] = (f16)x[(((size_t)b * NC + c) * NH + h) * NW + w];
  row[64] = (f16)((float)w / (float)(NW - 1));
#pragma unroll
  for (int c = 65; c < 96; ++c) row[c] = (f16)0.0f;
}

// bias + LayerNorm + exact GELU over 64-wide rows, write f16
__global__ void k_mlp_post(const float* pre, const float* bias, const float* g,
                           const float* be, f16* out, int npts) {
  int p = blockIdx.x * blockDim.x + threadIdx.x;
  if (p >= npts) return;
  const float* r = pre + (size_t)p * 64;
  float v[64], s = 0.0f, ss = 0.0f;
#pragma unroll
  for (int i = 0; i < 64; ++i) {
    float t = r[i] + bias[i];
    v[i] = t; s += t; ss += t * t;
  }
  float mu  = s * (1.0f / 64.0f);
  float var = ss * (1.0f / 64.0f) - mu * mu;
  float inv = rsqrtf(var + EPSV);
  f16* o = out + (size_t)p * 64;
#pragma unroll
  for (int i = 0; i < 64; ++i)
    o[i] = (f16)gelu_f(g[i] * (v[i] - mu) * inv + be[i]);
}

// wb epilogue: +bias, split re/im, write (b,h,w,64pad) and (b,h,m,192) layouts
__global__ void k_wb_epi(const float* Ow, const float* b2,
                         f16* r_wm, f16* i_wm, f16* r_mw, f16* i_mw) {
  int p = blockIdx.x * blockDim.x + threadIdx.x;
  if (p >= NB * NH * NW) return;
  int w = p % NW, h = (p / NW) % NH, b = p / (NH * NW);
  const float* r = Ow + (size_t)p * 96;
  size_t bh = (size_t)b * NH + h;
#pragma unroll 4
  for (int m = 0; m < 48; ++m) {
    float re = r[m] + b2[m];
    float im = r[48 + m] + b2[48 + m];
    r_wm[(size_t)p * 64 + m] = (f16)re;
    i_wm[(size_t)p * 64 + m] = (f16)im;
    size_t t = (bh * 48 + m) * NW + w;
    r_mw[t] = (f16)re;
    i_mw[t] = (f16)im;
  }
#pragma unroll
  for (int m = 48; m < 64; ++m) {
    r_wm[(size_t)p * 64 + m] = (f16)0.0f;
    i_wm[(size_t)p * 64 + m] = (f16)0.0f;
  }
}

// Xh[(b,m,h), 0:96] = {t.real[b,:,h,m], hc[h], zeros}; also t16 in (b,m,c,h)
__global__ void k_build_xh(const float* tr, const float* ti,
                           f16* Xh, f16* tr16, f16* ti16) {
  int q = blockIdx.x * blockDim.x + threadIdx.x;
  if (q >= NB * NM2 * NH) return;
  int h = q % NH, m = (q / NH) % NM2, b = q / (NM2 * NH);
  f16* row = Xh + (size_t)q * 96;
#pragma unroll 4
  for (int c = 0; c < 64; ++c) {
    size_t src = (((size_t)b * NC + c) * NH + h) * NM2 + m;
    float re = tr[src], im = ti[src];
    row[c] = (f16)re;
    size_t dst = (((size_t)b * NM2 + m) * NC + c) * NH + h;
    tr16[dst] = (f16)re;
    ti16[dst] = (f16)im;
  }
  row[64] = (f16)((float)h / (float)(NH - 1));
#pragma unroll
  for (int c = 65; c < 96; ++c) row[c] = (f16)0.0f;
}

// hb epilogue: +bias, split re/im, write (b,m,h,64pad) and (b,m,n,192) layouts
__global__ void k_hb_epi(const float* Oh, const float* b2,
                         f16* r_hn, f16* i_hn, f16* r_nh, f16* i_nh) {
  int q = blockIdx.x * blockDim.x + threadIdx.x;
  if (q >= NB * NM2 * NH) return;
  int h = q % NH, m = (q / NH) % NM2, b = q / (NM2 * NH);
  const float* r = Oh + (size_t)q * 96;
#pragma unroll 4
  for (int n = 0; n < 48; ++n) {
    float re = r[n] + b2[n];
    float im = r[48 + n] + b2[48 + n];
    r_hn[(size_t)q * 64 + n] = (f16)re;
    i_hn[(size_t)q * 64 + n] = (f16)im;
    size_t t = (((size_t)b * NM2 + m) * NM1 + n) * NH + h;
    r_nh[t] = (f16)re;
    i_nh[t] = (f16)im;
  }
#pragma unroll
  for (int n = 48; n < 64; ++n) {
    r_hn[(size_t)q * 64 + n] = (f16)0.0f;
    i_hn[(size_t)q * 64 + n] = (f16)0.0f;
  }
}

// t2 f32 (x,y,16,64) -> f16, zeroing pad rows b>=8
__global__ void k_t2cast(const float* t2r, const float* t2i, f16* r16, f16* i16) {
  int i = blockIdx.x * blockDim.x + threadIdx.x;
  if (i >= NM1 * NM2 * 16 * 64) return;
  int bp = (i >> 6) & 15;
  if (bp < 8) { r16[i] = (f16)t2r[i]; i16[i] = (f16)t2i[i]; }
  else        { r16[i] = (f16)0.0f;   i16[i] = (f16)0.0f;   }
}

// freq (o,i,x,y) f32 -> (x,y,o,i) f16
__global__ void k_freq_cast(const float* fr, const float* fi, f16* fr16, f16* fi16) {
  int i = blockIdx.x * blockDim.x + threadIdx.x;
  if (i >= NM1 * NM2 * 64 * 64) return;
  int ii = i & 63;
  int o  = (i >> 6) & 63;
  int y  = (i >> 12) % NM2;
  int x  = i / (64 * 64 * NM2);
  size_t src = (((size_t)o * 64 + ii) * NM1 + x) * NM2 + y;
  fr16[i] = (f16)fr[src];
  fi16[i] = (f16)fi[src];
}

// proc f32 (x,y,16,64) -> f16 (b,y,o,xpad64), zero x>=48
__global__ void k_proc_cast(const float* pr, const float* pi, f16* pr16, f16* pi16) {
  int i = blockIdx.x * blockDim.x + threadIdx.x;
  if (i >= 8 * NM2 * 64 * 64) return;
  int xp = i & 63;
  int o  = (i >> 6) & 63;
  int y  = (i >> 12) % NM2;
  int b  = i / (64 * 64 * NM2);
  if (xp < 48) {
    size_t src = (((size_t)xp * NM2 + y) * 16 + b) * 64 + o;
    pr16[i] = (f16)pr[src];
    pi16[i] = (f16)pi[src];
  } else {
    pr16[i] = (f16)0.0f;
    pi16[i] = (f16)0.0f;
  }
}

// rec_h f32 (b,m,o,h) -> f16 (b,h,o,mpad64), zero m>=48
__global__ void k_rh_cast(const float* rr, const float* ri, f16* r16, f16* i16) {
  int i = blockIdx.x * blockDim.x + threadIdx.x;
  if (i >= NB * NH * 64 * 64) return;
  int mp = i & 63;
  int o  = (i >> 6) & 63;
  int h  = (i >> 12) % NH;
  int b  = i / (64 * 64 * NH);
  if (mp < 48) {
    size_t src = (((size_t)b * NM2 + mp) * NO + o) * NH + h;
    r16[i] = (f16)rr[src];
    i16[i] = (f16)ri[src];
  } else {
    r16[i] = (f16)0.0f;
    i16[i] = (f16)0.0f;
  }
}

// final: +mixer_b, channel LayerNorm, GELU, + (shortcut + sc_b), GELU -> (B,O,H,W)
__global__ void k_final(const float* z, const float* sc, const float* mixb,
                        const float* ng, const float* nb, const float* scb,
                        float* out) {
  int p = blockIdx.x * blockDim.x + threadIdx.x;
  if (p >= NB * NH * NW) return;
  int w = p % NW, h = (p / NW) % NH, b = p / (NH * NW);
  const float* zr = z + (size_t)p * 64;
  const float* sr = sc + (size_t)p * 64;
  float v[64], s = 0.0f, ss = 0.0f;
#pragma unroll
  for (int o = 0; o < 64; ++o) {
    float t = zr[o] + mixb[o];
    v[o] = t; s += t; ss += t * t;
  }
  float mu  = s * (1.0f / 64.0f);
  float var = ss * (1.0f / 64.0f) - mu * mu;
  float inv = rsqrtf(var + EPSV);
#pragma unroll
  for (int o = 0; o < 64; ++o) {
    float t = gelu_f(ng[o] * (v[o] - mu) * inv + nb[o]);
    t = gelu_f(t + sr[o] + scb[o]);
    out[(((size_t)b * NO + o) * NH + h) * NW + w] = t;
  }
}

// ---------------------------------------------------------------------------
// Host side
// ---------------------------------------------------------------------------
static void launch_gemm(hipStream_t s, const f16* A, const f16* Bt, float* D,
                        int M, int N, int K, int lda, int ldb, int ldc,
                        long long sA1, long long sA2, long long sB1, long long sB2,
                        long long sD1, long long sD2, int B1, int B2,
                        float alpha, float beta) {
  GemmP p{A, Bt, D, M, N, K, lda, ldb, ldc, sA1, sA2, sB1, sB2, sD1, sD2, B2, alpha, beta};
  if ((M % 128 == 0) && (N % 32 == 0)) {
    dim3 g((unsigned)(N / 32), (unsigned)(M / 128), (unsigned)(B1 * B2));
    wmma_gemm_mlp<<<g, 128, 0, s>>>(p);
  } else if ((M % 32 == 0) && (N % 32 == 0)) {
    dim3 g((unsigned)(N / 32), (unsigned)(M / 32), (unsigned)(B1 * B2));
    wmma_gemm32<<<g, 32, 0, s>>>(p);
  } else {
    dim3 g((unsigned)(N / 16), (unsigned)(M / 16), (unsigned)(B1 * B2));
    wmma_gemm16<<<g, 32, 0, s>>>(p);
  }
}

extern "C" void kernel_launch(void* const* d_in, const int* in_sizes, int n_in,
                              void* d_out, int out_size, void* d_ws, size_t ws_size,
                              hipStream_t stream) {
  const float* x    = (const float*)d_in[0];
  const float* wW0  = (const float*)d_in[1];
  const float* wb0  = (const float*)d_in[2];
  const float* wg0  = (const float*)d_in[3];
  const float* wbe0 = (const float*)d_in[4];
  const float* wW1  = (const float*)d_in[5];
  const float* wb1  = (const float*)d_in[6];
  const float* wg1  = (const float*)d_in[7];
  const float* wbe1 = (const float*)d_in[8];
  const float* wW2  = (const float*)d_in[9];
  const float* wb2  = (const float*)d_in[10];
  const float* hW0  = (const float*)d_in[11];
  const float* hb0  = (const float*)d_in[12];
  const float* hg0  = (const float*)d_in[13];
  const float* hbe0 = (const float*)d_in[14];
  const float* hW1  = (const float*)d_in[15];
  const float* hb1  = (const float*)d_in[16];
  const float* hg1  = (const float*)d_in[17];
  const float* hbe1 = (const float*)d_in[18];
  const float* hW2  = (const float*)d_in[19];
  const float* hb2  = (const float*)d_in[20];
  const float* frq  = (const float*)d_in[21];
  const float* fiq  = (const float*)d_in[22];
  const float* mixw = (const float*)d_in[23];
  const float* mixb = (const float*)d_in[24];
  const float* ngv  = (const float*)d_in[25];
  const float* nbv  = (const float*)d_in[26];
  const float* scw  = (const float*)d_in[27];
  const float* scbv = (const float*)d_in[28];

  const size_t P = (size_t)NB * NH * NW;   // 294912 W-MLP points
  const size_t Q = (size_t)NB * NM2 * NH;  //  73728 H-MLP points

  char* ws = (char*)d_ws;
  size_t off = 0;
  auto alloc = [&](size_t bytes) -> void* {
    void* p = ws + off;
    off += (bytes + 255) & ~(size_t)255;
    return p;
  };
  // staged f16 weights
  f16* w0t = (f16*)alloc(64 * 96 * 2);
  f16* w1t = (f16*)alloc(64 * 64 * 2);
  f16* w2t = (f16*)alloc(96 * 64 * 2);
  f16* h0t = (f16*)alloc(64 * 96 * 2);
  f16* h1t = (f16*)alloc(64 * 64 * 2);
  f16* h2t = (f16*)alloc(96 * 64 * 2);
  f16* mx16  = (f16*)alloc(64 * 64 * 2);
  f16* scw16 = (f16*)alloc(64 * 96 * 2);
  // big staging (total ~0.98 GB)
  f16*   Xw   = (f16*)alloc(P * 96 * 2);
  float* Hf   = (float*)alloc(P * 64 * 4);          // reused by H-MLP
  f16*   Hht  = (f16*)alloc(P * 64 * 2);            // reused by H-MLP
  float* Ow   = (float*)alloc(P * 96 * 4);          // reused as Oh
  f16*   wbr_wm = (f16*)alloc(P * 64 * 2);
  f16*   wbi_wm = (f16*)alloc(P * 64 * 2);
  f16*   wbr_mw = (f16*)alloc((size_t)NB * NH * 48 * NW * 2);
  f16*   wbi_mw = (f16*)alloc((size_t)NB * NH * 48 * NW * 2);
  f16*   x16  = (f16*)alloc((size_t)NB * NC * NH * NW * 2);
  float* t_r  = (float*)alloc((size_t)NB * NC * NH * 48 * 4);
  float* t_i  = (float*)alloc((size_t)NB * NC * NH * 48 * 4);
  f16*   tr16 = (f16*)alloc((size_t)NB * 48 * 64 * NH * 2);
  f16*   ti16 = (f16*)alloc((size_t)NB * 48 * 64 * NH * 2);
  f16*   Xh   = (f16*)alloc(Q * 96 * 2);
  f16*   hbr_hn = (f16*)alloc(Q * 64 * 2);
  f16*   hbi_hn = (f16*)alloc(Q * 64 * 2);
  f16*   hbr_nh = (f16*)alloc((size_t)NB * 48 * 48 * NH * 2);
  f16*   hbi_nh = (f16*)alloc((size_t)NB * 48 * 48 * NH * 2);
  float* t2r  = (float*)alloc((size_t)48 * 48 * 16 * 64 * 4);
  float* t2i  = (float*)alloc((size_t)48 * 48 * 16 * 64 * 4);
  f16*   t2r16 = (f16*)alloc((size_t)48 * 48 * 16 * 64 * 2);
  f16*   t2i16 = (f16*)alloc((size_t)48 * 48 * 16 * 64 * 2);
  f16*   fr16 = (f16*)alloc((size_t)48 * 48 * 64 * 64 * 2);
  f16*   fi16 = (f16*)alloc((size_t)48 * 48 * 64 * 64 * 2);
  float* prcr = (float*)alloc((size_t)48 * 48 * 16 * 64 * 4);
  float* prci = (float*)alloc((size_t)48 * 48 * 16 * 64 * 4);
  f16*   pr16 = (f16*)alloc((size_t)8 * 48 * 64 * 64 * 2);
  f16*   pi16 = (f16*)alloc((size_t)8 * 48 * 64 * 64 * 2);
  float* rhr  = (float*)alloc((size_t)8 * 48 * 64 * 192 * 4);
  float* rhi  = (float*)alloc((size_t)8 * 48 * 64 * 192 * 4);
  f16*   rh16r = (f16*)alloc((size_t)8 * 192 * 64 * 64 * 2);
  f16*   rh16i = (f16*)alloc((size_t)8 * 192 * 64 * 64 * 2);
  float* ybuf = (float*)alloc(P * 64 * 4);
  f16*   y16  = (f16*)alloc(P * 64 * 2);
  float* zbuf = (float*)alloc(P * 64 * 4);
  float* scbf = (float*)alloc(P * 64 * 4);
  (void)ws_size; (void)n_in; (void)in_sizes; (void)out_size;

  const int T = 256;
  auto G = [](size_t n) { return dim3((unsigned)((n + 255) / 256)); };

  // weights -> f16 (transposed to N x Kpad, K-contiguous)
  k_tcast<<<G(64 * 96), T, 0, stream>>>(wW0, w0t, 64, 96, 65, 64);
  k_tcast<<<G(64 * 64), T, 0, stream>>>(wW1, w1t, 64, 64, 64, 64);
  k_tcast<<<G(96 * 64), T, 0, stream>>>(wW2, w2t, 96, 64, 64, 96);
  k_tcast<<<G(64 * 96), T, 0, stream>>>(hW0, h0t, 64, 96, 65, 64);
  k_tcast<<<G(64 * 64), T, 0, stream>>>(hW1, h1t, 64, 64, 64, 64);
  k_tcast<<<G(96 * 64), T, 0, stream>>>(hW2, h2t, 96, 64, 64, 96);
  k_ccast<<<G(64 * 64), T, 0, stream>>>(mixw, mx16, 64, 64, 64, 64);
  k_ccast<<<G(64 * 96), T, 0, stream>>>(scw, scw16, 64, 96, 64, 64);
  k_build_xw<<<G(P), T, 0, stream>>>(x, Xw);
  k_cast<<<G((size_t)NB * NC * NH * NW), T, 0, stream>>>(x, x16, NB * NC * NH * NW);

  // ---- W-basis MLP: (P x 96) -> 64 -> 64 -> 96 ----
  launch_gemm(stream, Xw, w0t, Hf, (int)P, 64, 96, 96, 96, 64, 0,0,0,0,0,0, 1,1, 1.f, 0.f);
  k_mlp_post<<<G(P), T, 0, stream>>>(Hf, wb0, wg0, wbe0, Hht, (int)P);
  launch_gemm(stream, Hht, w1t, Hf, (int)P, 64, 64, 64, 64, 64, 0,0,0,0,0,0, 1,1, 1.f, 0.f);
  k_mlp_post<<<G(P), T, 0, stream>>>(Hf, wb1, wg1, wbe1, Hht, (int)P);
  launch_gemm(stream, Hht, w2t, Ow, (int)P, 96, 64, 64, 64, 96, 0,0,0,0,0,0, 1,1, 1.f, 0.f);
  k_wb_epi<<<G(P), T, 0, stream>>>(Ow, wb2, wbr_wm, wbi_wm, wbr_mw, wbi_mw);

  // ---- t[b,c,h,m] = sum_w x[b,c,h,w] * wb[b,h,w,m]  (per (b,h): 64x48x192) ----
  launch_gemm(stream, x16, wbr_mw, t_r, 64, 48, 192, NH*NW, NW, NH*48,
              (long long)NC*NH*NW, NW, (long long)NH*48*NW, 48LL*NW,
              (long long)NC*NH*48, 48, NB, NH, 1.f, 0.f);
  launch_gemm(stream, x16, wbi_mw, t_i, 64, 48, 192, NH*NW, NW, NH*48,
              (long long)NC*NH*NW, NW, (long long)NH*48*NW, 48LL*NW,
              (long long)NC*NH*48, 48, NB, NH, 1.f, 0.f);
  k_build_xh<<<G(Q), T, 0, stream>>>(t_r, t_i, Xh, tr16, ti16);

  // ---- H-basis MLP (reuse Hf/Hht/Ow) ----
  launch_gemm(stream, Xh, h0t, Hf, (int)Q, 64, 96, 96, 96, 64, 0,0,0,0,0,0, 1,1, 1.f, 0.f);
  k_mlp_post<<<G(Q), T, 0, stream>>>(Hf, hb0, hg0, hbe0, Hht, (int)Q);
  launch_gemm(stream, Hht, h1t, Hf, (int)Q, 64, 64, 64, 64, 64, 0,0,0,0,0,0, 1,1, 1.f, 0.f);
  k_mlp_post<<<G(Q), T, 0, stream>>>(Hf, hb1, hg1, hbe1, Hht, (int)Q);
  launch_gemm(stream, Hht, h2t, Ow, (int)Q, 96, 64, 64, 64, 96, 0,0,0,0,0,0, 1,1, 1.f, 0.f);
  k_hb_epi<<<G(Q), T, 0, stream>>>(Ow, hb2, hbr_hn, hbi_hn, hbr_nh, hbi_nh);

  // ---- t2[n,m,b,c] = sum_h hb[b,m,n,h] * t[b,m,c,h]  (complex, 4 GEMMs) ----
  {
    const long long aB = 48LL*48*NH, aM = 48LL*NH;
    const long long bB = 48LL*64*NH, bM = 64LL*NH;
    const long long dB = 64, dM = 16LL*64;
    launch_gemm(stream, hbr_nh, tr16, t2r, 48, 64, 192, NH, NH, 48*16*64, aB,aM,bB,bM,dB,dM, NB,48,  1.f, 0.f);
    launch_gemm(stream, hbi_nh, ti16, t2r, 48, 64, 192, NH, NH, 48*16*64, aB,aM,bB,bM,dB,dM, NB,48, -1.f, 1.f);
    launch_gemm(stream, hbr_nh, ti16, t2i, 48, 64, 192, NH, NH, 48*16*64, aB,aM,bB,bM,dB,dM, NB,48,  1.f, 0.f);
    launch_gemm(stream, hbi_nh, tr16, t2i, 48, 64, 192, NH, NH, 48*16*64, aB,aM,bB,bM,dB,dM, NB,48,  1.f, 1.f);
  }
  k_t2cast<<<G((size_t)48*48*16*64), T, 0, stream>>>(t2r, t2i, t2r16, t2i16);
  k_freq_cast<<<G((size_t)48*48*64*64), T, 0, stream>>>(frq, fiq, fr16, fi16);

  // ---- proc[x,y,b,o] = sum_i t2[x,y,b,i] * Wf[o,i,x,y]  (per (x,y): 16x64x64) ----
  {
    const long long aX = 48LL*16*64, aY = 16LL*64;
    const long long bX = 48LL*64*64, bY = 64LL*64;
    launch_gemm(stream, t2r16, fr16, prcr, 16, 64, 64, 64, 64, 64, aX,aY,bX,bY,aX,aY, 48,48,  1.f, 0.f);
    launch_gemm(stream, t2i16, fi16, prcr, 16, 64, 64, 64, 64, 64, aX,aY,bX,bY,aX,aY, 48,48, -1.f, 1.f);
    launch_gemm(stream, t2r16, fi16, prci, 16, 64, 64, 64, 64, 64, aX,aY,bX,bY,aX,aY, 48,48,  1.f, 0.f);
    launch_gemm(stream, t2i16, fr16, prci, 16, 64, 64, 64, 64, 64, aX,aY,bX,bY,aX,aY, 48,48,  1.f, 1.f);
  }
  k_proc_cast<<<G((size_t)8*48*64*64), T, 0, stream>>>(prcr, prci, pr16, pi16);

  // ---- rec_h[b,m,o,h] = sum_n proc[b,m,o,n] * conj(hb)[b,m,h,n]  (64x192x64) ----
  {
    const long long aB = 48LL*64*64, aM = 64LL*64;
    const long long bB = 48LL*NH*64, bM = (long long)NH*64;
    const long long dB = 48LL*64*NH, dM = 64LL*NH;
    launch_gemm(stream, pr16, hbr_hn, rhr, 64, 192, 64, 64, 64, NH, aB,aM,bB,bM,dB,dM, NB,48,  1.f, 0.f);
    launch_gemm(stream, pi16, hbi_hn, rhr, 64, 192, 64, 64, 64, NH, aB,aM,bB,bM,dB,dM, NB,48,  1.f, 1.f);
    launch_gemm(stream, pi16, hbr_hn, rhi, 64, 192, 64, 64, 64, NH, aB,aM,bB,bM,dB,dM, NB,48,  1.f, 0.f);
    launch_gemm(stream, pr16, hbi_hn, rhi, 64, 192, 64, 64, 64, NH, aB,aM,bB,bM,dB,dM, NB,48, -1.f, 1.f);
  }
  k_rh_cast<<<G((size_t)NB*NH*64*64), T, 0, stream>>>(rhr, rhi, rh16r, rh16i);

  // ---- y[b,h,w,o] = Re(rec)/HW = (rhr.wbr + rhi.wbi)/HW  (per (b,h): 192x64x64) ----
  {
    const float a = 1.0f / (float)(NH * NW);
    const long long aB = (long long)NH*NW*64, aH = (long long)NW*64;
    const long long bB = (long long)NH*64*64, bH = 64LL*64;
    launch_gemm(stream, wbr_wm, rh16r, ybuf, 192, 64, 64, 64, 64, 64, aB,aH,bB,bH,aB,aH, NB,NH, a, 0.f);
    launch_gemm(stream, wbi_wm, rh16i, ybuf, 192, 64, 64, 64, 64, 64, aB,aH,bB,bH,aB,aH, NB,NH, a, 1.f);
  }
  k_cast<<<G(P * 64), T, 0, stream>>>(ybuf, y16, (int)(P * 64));

  // ---- mixer + shortcut GEMMs (per (b,h): 192x64) ----
  {
    const long long aB = (long long)NH*NW*64, aH = (long long)NW*64;
    launch_gemm(stream, y16, mx16, zbuf, 192, 64, 64, 64, 64, 64, aB,aH,0,0,aB,aH, NB,NH, 1.f, 0.f);
    const long long xB = (long long)NH*NW*96, xH = (long long)NW*96;
    launch_gemm(stream, Xw, scw16, scbf, 192, 64, 96, 96, 96, 64, xB,xH,0,0,aB,aH, NB,NH, 1.f, 0.f);
  }

  // ---- channel-LN + GELU + shortcut + GELU -> (B,O,H,W) ----
  k_final<<<G(P), T, 0, stream>>>(zbuf, scbf, mixb, ngv, nbv, scbv, (float*)d_out);
}
